// WindowSelfAttention_1219770712154
// MI455X (gfx1250) — compile-verified, matrix operand order checked by reading, not run on hardware
//
#include <hip/hip_runtime.h>

typedef _Float16 half_t;
typedef __attribute__((ext_vector_type(16))) _Float16 v16h;
typedef __attribute__((ext_vector_type(8)))  _Float16 v8h;
typedef __attribute__((ext_vector_type(8)))  float    v8f;

#define DIM   512
#define WS2   64
#define HEADS 16
#define HD    32
#define NWIN  1024          /* 16 batches * 8 * 8 windows */
#define EPSV  1e-5f
#define SCL   0.17677669529663687f  /* 1/sqrt(32) */
#define ASTR  40            /* padded LDS stride (halves) for 64x32 A tiles */

// ---------------------------------------------------------------- WMMA helpers
__device__ __forceinline__ v8f wmma_fp16(v16h a, v16h b, v8f c) {
  return __builtin_amdgcn_wmma_f32_16x16x32_f16(false, a, false, b, (short)0, c,
                                                false, false);
}

// Scheduling hint: issue B vmem reads, then all 8 A ds reads, then 4 WMMAs.
__device__ __forceinline__ void sched_gemm_group() {
#if __has_builtin(__builtin_amdgcn_sched_group_barrier)
  __builtin_amdgcn_sched_group_barrier(0x020, 2, 0);  // 2x VMEM read  (B frag)
  __builtin_amdgcn_sched_group_barrier(0x100, 8, 0);  // 8x DS read    (A frags)
  __builtin_amdgcn_sched_group_barrier(0x008, 4, 0);  // 4x WMMA
#endif
}
__device__ __forceinline__ void sched_lds_gemm_group() {
#if __has_builtin(__builtin_amdgcn_sched_group_barrier)
  __builtin_amdgcn_sched_group_barrier(0x100, 10, 0); // 10x DS read (B + A)
  __builtin_amdgcn_sched_group_barrier(0x008, 4, 0);  // 4x WMMA
#endif
}

// Per ISA 7.12.2, a lane's 16 fragment halves (A row m / B column n) are two
// contiguous 8-half runs: k = hi8+0..7 and k = 16+hi8+0..7.  `p` must point at
// (row_base + hi8) and be 16B aligned -> two b128 loads.
__device__ __forceinline__ v16h frag_ld(const half_t* p) {
  const v8h lo = *(const v8h*)(p);
  const v8h hi = *(const v8h*)(p + 16);
  v16h r;
#pragma unroll
  for (int i = 0; i < 8; ++i) { r[i] = lo[i]; r[8 + i] = hi[i]; }
  return r;
}

// ------------------------------------------------ CDNA5 async copy -> LDS
// GLOBAL_LOAD_ASYNC_TO_LDS_B128 (ISA 15.18.3, ASYNCcnt). LDS dest address is
// the low 32 bits of the flat shared pointer (LDS aperture truncation).
__device__ __forceinline__ unsigned lds_addr(const void* p) {
  return (unsigned)(unsigned long long)(uintptr_t)p;
}
__device__ __forceinline__ void async_copy16(unsigned lds, const void* g) {
  asm volatile("global_load_async_to_lds_b128 %0, %1, off"
               :: "v"(lds), "v"((unsigned long long)(uintptr_t)g)
               : "memory");
}
__device__ __forceinline__ void wait_async0() {
  asm volatile("s_wait_asynccnt 0" ::: "memory");
}

// --------------------------------------------------- kernel 1: prep (weights)
__global__ void prep_kernel(const float* __restrict__ qk_w,
                            const float* __restrict__ v_w,
                            const float* __restrict__ proj_w,
                            const float* __restrict__ g1, const float* __restrict__ be1,
                            const float* __restrict__ m1, const float* __restrict__ va1,
                            const float* __restrict__ g2, const float* __restrict__ be2,
                            const float* __restrict__ m2, const float* __restrict__ va2,
                            half_t* __restrict__ wqk, half_t* __restrict__ wv,
                            half_t* __restrict__ wpj,
                            float* __restrict__ bn1s, float* __restrict__ bn1h,
                            float* __restrict__ bn2s, float* __restrict__ bn2h) {
  int i = blockIdx.x * blockDim.x + threadIdx.x;
  const int NQK = 2 * DIM * DIM;   // 524288
  const int NV  = DIM * DIM;       // 262144
  if (i < NQK) { wqk[i] = (half_t)qk_w[i]; return; }
  i -= NQK;
  if (i < NV) { wv[i] = (half_t)v_w[i]; return; }
  i -= NV;
  if (i < NV) { wpj[i] = (half_t)proj_w[i]; return; }
  i -= NV;
  if (i < DIM) {
    const float s = g1[i] * rsqrtf(va1[i] + EPSV);
    bn1s[i] = s; bn1h[i] = be1[i] - m1[i] * s; return;
  }
  i -= DIM;
  if (i < DIM) {
    const float s = g2[i] * rsqrtf(va2[i] + EPSV);
    bn2s[i] = s; bn2h[i] = be2[i] - m2[i] * s; return;
  }
}

// -------------------------------------------- kernel 2: fused BN1 + QKV GEMMs
// Block = 256 thr (8 waves): one window, 128 of 1536 output cols.
// cols [0,512)=Q, [512,1024)=K, [1024,1536)=V (V stored transposed per head).
// B fragments come straight from global f16 weights (two b128 per K-step).
__global__ void __launch_bounds__(256)
qkv_kernel(const float* __restrict__ x,
           const half_t* __restrict__ wqk, const half_t* __restrict__ wv,
           const float* __restrict__ bn1s, const float* __restrict__ bn1h,
           const float* __restrict__ qk_b, const float* __restrict__ v_b,
           half_t* __restrict__ qbuf, half_t* __restrict__ kbuf,
           half_t* __restrict__ vbuf) {
  __shared__ __align__(16) half_t As[2][WS2 * ASTR];   // double-buffered A tile

  const int w     = blockIdx.x / 12;
  const int nbase = (blockIdx.x % 12) * 128;
  const int tid = threadIdx.x, lane = tid & 31, wid = tid >> 5;  // wid = N tile
  const int nl = lane & 15, hi8 = (lane >> 4) << 3;

  const int b = w >> 6, wi = w & 63;
  const int row0 = (wi >> 3) << 3, col0 = (wi & 7) << 3;

  // per-lane output column + weight row + routing (resolved once)
  const int ng = nbase + wid * 16 + nl;
  const half_t* wrow; float bias; half_t* dst; int base, strideT;
  if (ng < 2 * DIM) {
    wrow = wqk + (size_t)ng * DIM;
    bias = qk_b[ng];
    const int nn = (ng < DIM) ? ng : ng - DIM;
    dst = (ng < DIM) ? qbuf : kbuf;
    base = (((w * HEADS) + (nn >> 5)) * WS2) * HD + (nn & 31);   // [t][hd]
    strideT = HD;
  } else {
    const int nn = ng - 2 * DIM;
    wrow = wv + (size_t)nn * DIM;
    bias = v_b[nn];
    dst  = vbuf;
    base = (((w * HEADS) + (nn >> 5)) * HD + (nn & 31)) * WS2;   // [hd][t]
    strideT = 1;
  }

  // A-tile staging: token-fastest for coalesced x reads (4x32B runs per wave)
  auto stage = [&](int ks, half_t* buf) {
    for (int i = tid; i < WS2 * 32; i += 256) {
      const int t = i & 63, kk = i >> 6;
      const int c = (ks << 5) + kk;
      const int pr = row0 + (t >> 3), pc = col0 + (t & 7);
      const float xv = x[((b * DIM + c) << 12) + (pr << 6) + pc];
      buf[t * ASTR + kk] = (half_t)(xv * bn1s[c] + bn1h[c]);
    }
  };

  v8f acc[4] = {};
  stage(0, As[0]);
  for (int ks = 0; ks < 16; ++ks) {
    __syncthreads();
    if (ks < 15) stage(ks + 1, As[(ks + 1) & 1]);
    const half_t* Ab = As[ks & 1] + nl * ASTR + hi8;
    const v16h bf  = frag_ld(wrow + (ks << 5) + hi8);   // global b128 x2
    const v16h af0 = frag_ld(Ab);                       // all A frags first:
    const v16h af1 = frag_ld(Ab + 16 * ASTR);           // lets ds loads pipeline
    const v16h af2 = frag_ld(Ab + 32 * ASTR);
    const v16h af3 = frag_ld(Ab + 48 * ASTR);
    acc[0] = wmma_fp16(af0, bf, acc[0]);
    acc[1] = wmma_fp16(af1, bf, acc[1]);
    acc[2] = wmma_fp16(af2, bf, acc[2]);
    acc[3] = wmma_fp16(af3, bf, acc[3]);
    sched_gemm_group();
  }

#pragma unroll
  for (int m = 0; m < 4; ++m)
#pragma unroll
    for (int r = 0; r < 8; ++r) {
      const int t = m * 16 + r + hi8;
      dst[base + t * strideT] = (half_t)(acc[m][r] + bias);
    }
}

// ------------------------------ kernel 3: per (window, head) attention, WMMA
__global__ void __launch_bounds__(128)
attn_kernel(const half_t* __restrict__ qbuf, const half_t* __restrict__ kbuf,
            const half_t* __restrict__ vbuf, const float* __restrict__ bias_table,
            half_t* __restrict__ ao) {
  __shared__ __align__(16) half_t qs[WS2 * HD];     // [token][hd]
  __shared__ __align__(16) half_t ksm[WS2 * HD];    // [token][hd]
  __shared__ __align__(16) half_t vs[HD * WS2];     // [hd][token] (transposed)
  __shared__ __align__(16) float  Sf[WS2 * WS2];
  __shared__ __align__(16) half_t Ps[WS2 * WS2];

  const int w = blockIdx.x >> 4, h = blockIdx.x & 15;
  const int tid = threadIdx.x, lane = tid & 31, wid = tid >> 5;  // wid: 0..3
  const int nl = lane & 15, hi8 = (lane >> 4) << 3;

  // async-DMA q/k/v tiles (3 x 4KB contiguous) into LDS
  const size_t off = (size_t)(w * HEADS + h) * (WS2 * HD);
  const char* qg = (const char*)(qbuf + off);
  const char* kg = (const char*)(kbuf + off);
  const char* vg = (const char*)(vbuf + off);
  const unsigned ql = lds_addr(qs), kl = lds_addr(ksm), vl = lds_addr(vs);
  for (int i = tid; i < 256; i += 128) {            // 256 x 16B per array
    const int byo = i * 16;
    async_copy16(ql + byo, qg + byo);
    async_copy16(kl + byo, kg + byo);
    async_copy16(vl + byo, vg + byo);
  }
  wait_async0();
  __syncthreads();

  { // S = q @ k^T  (64x64, K=32). Wave handles N tile `wid`, loops 4 M tiles.
    const v16h bf  = frag_ld(ksm + (wid * 16 + nl) * HD + hi8); // k token = col
    const half_t* Qb = qs + nl * HD + hi8;
    const v16h af0 = frag_ld(Qb);
    const v16h af1 = frag_ld(Qb + 16 * HD);
    const v16h af2 = frag_ld(Qb + 32 * HD);
    const v16h af3 = frag_ld(Qb + 48 * HD);
    v8f acc[4] = {};
    acc[0] = wmma_fp16(af0, bf, acc[0]);
    acc[1] = wmma_fp16(af1, bf, acc[1]);
    acc[2] = wmma_fp16(af2, bf, acc[2]);
    acc[3] = wmma_fp16(af3, bf, acc[3]);
    sched_lds_gemm_group();
    const int kt = wid * 16 + nl;
#pragma unroll
    for (int m = 0; m < 4; ++m)
#pragma unroll
      for (int r = 0; r < 8; ++r) {
        const int qt = m * 16 + r + hi8;
        const int dr = (qt >> 3) - (kt >> 3) + 7;
        const int dc = (qt & 7) - (kt & 7) + 7;
        Sf[qt * WS2 + kt] = acc[m][r] * SCL + bias_table[(dr * 15 + dc) * HEADS + h];
      }
  }
  __syncthreads();

  if (tid < WS2) {  // row softmax
    const float* row = Sf + tid * WS2;
    float mx = -1e30f;
    for (int j = 0; j < WS2; ++j) mx = fmaxf(mx, row[j]);
    float sum = 0.f;
    for (int j = 0; j < WS2; ++j) sum += __expf(row[j] - mx);
    const float inv = 1.0f / sum;
    for (int j = 0; j < WS2; ++j)
      Ps[tid * WS2 + j] = (half_t)(__expf(row[j] - mx) * inv);
  }
  __syncthreads();

  // out = P @ V  (M=64 N=32 K=64). Wave handles M tile `wid`.
  v8f acc2[2] = {};
#pragma unroll
  for (int kst = 0; kst < 2; ++kst) {
    const v16h af = frag_ld(Ps + (wid * 16 + nl) * WS2 + kst * 32 + hi8);
#pragma unroll
    for (int nt = 0; nt < 2; ++nt) {  // V transposed: contiguous b128 pairs
      const v16h bf = frag_ld(vs + (nt * 16 + nl) * WS2 + kst * 32 + hi8);
      acc2[nt] = wmma_fp16(af, bf, acc2[nt]);
    }
  }
#pragma unroll
  for (int nt = 0; nt < 2; ++nt)
#pragma unroll
    for (int r = 0; r < 8; ++r) {
      const int t  = wid * 16 + r + hi8;
      const int hd = nt * 16 + nl;
      ao[((size_t)w * WS2 + t) * DIM + h * HD + hd] = (half_t)acc2[nt][r];
    }
}

// ------------------- kernel 4: proj GEMM + residual + BN2, NCHW fp32 output
__global__ void __launch_bounds__(256)
proj_kernel(const float* __restrict__ x, const half_t* __restrict__ ao,
            const half_t* __restrict__ wpj, const float* __restrict__ proj_b,
            const float* __restrict__ bn2s, const float* __restrict__ bn2h,
            float* __restrict__ out) {
  __shared__ __align__(16) half_t As[2][WS2 * ASTR];

  const int w = blockIdx.x >> 2;
  const int nbase = (blockIdx.x & 3) * 128;
  const int tid = threadIdx.x, lane = tid & 31, wid = tid >> 5;
  const int nl = lane & 15, hi8 = (lane >> 4) << 3;
  const int b = w >> 6, wi = w & 63;
  const int row0 = (wi >> 3) << 3, col0 = (wi & 7) << 3;

  const int ch = nbase + wid * 16 + nl;                 // output channel
  const half_t* wrow = wpj + (size_t)ch * DIM;

  const unsigned As_l = lds_addr(As);
  // async-DMA stage: 64 rows x 64B of ao -> padded LDS tile (4 x b128 / row)
  auto stage = [&](int ks, int bufi) {
    const int t = tid >> 2, part = tid & 3;             // 256 chunks of 16B
    const char* g = (const char*)(ao + ((size_t)w * WS2 + t) * DIM + (ks << 5))
                    + part * 16;
    const unsigned l = As_l + (unsigned)bufi * (WS2 * ASTR * 2)
                     + (unsigned)(t * ASTR * 2 + part * 16);
    async_copy16(l, g);
  };

  v8f acc[4] = {};
  stage(0, 0);
  for (int ks = 0; ks < 16; ++ks) {
    wait_async0();
    __syncthreads();
    if (ks < 15) stage(ks + 1, (ks + 1) & 1);
    const half_t* Ab = As[ks & 1] + nl * ASTR + hi8;
    const v16h bf  = frag_ld(wrow + (ks << 5) + hi8);
    const v16h af0 = frag_ld(Ab);
    const v16h af1 = frag_ld(Ab + 16 * ASTR);
    const v16h af2 = frag_ld(Ab + 32 * ASTR);
    const v16h af3 = frag_ld(Ab + 48 * ASTR);
    acc[0] = wmma_fp16(af0, bf, acc[0]);
    acc[1] = wmma_fp16(af1, bf, acc[1]);
    acc[2] = wmma_fp16(af2, bf, acc[2]);
    acc[3] = wmma_fp16(af3, bf, acc[3]);
    sched_gemm_group();
  }

  const float pb = proj_b[ch], s2 = bn2s[ch], h2 = bn2h[ch];
#pragma unroll
  for (int m = 0; m < 4; ++m)
#pragma unroll
    for (int r = 0; r < 8; ++r) {
      const int t  = m * 16 + r + hi8;
      const int pr = row0 + (t >> 3), pc = col0 + (t & 7);
      const int idx = ((b * DIM + ch) << 12) + (pr << 6) + pc;
      const float res = x[idx] + acc[m][r] + pb;  // residual
      out[idx] = res * s2 + h2;                   // BN2
    }
}

// ----------------------------------------------------------------- launcher
extern "C" void kernel_launch(void* const* d_in, const int* in_sizes, int n_in,
                              void* d_out, int out_size, void* d_ws, size_t ws_size,
                              hipStream_t stream) {
  const float* x    = (const float*)d_in[0];
  const float* g1   = (const float*)d_in[1];
  const float* be1  = (const float*)d_in[2];
  const float* m1   = (const float*)d_in[3];
  const float* va1  = (const float*)d_in[4];
  const float* qk_w = (const float*)d_in[5];
  const float* qk_b = (const float*)d_in[6];
  const float* v_w  = (const float*)d_in[7];
  const float* v_b  = (const float*)d_in[8];
  const float* pw   = (const float*)d_in[9];
  const float* pb   = (const float*)d_in[10];
  const float* bt   = (const float*)d_in[11];
  const float* g2   = (const float*)d_in[12];
  const float* be2  = (const float*)d_in[13];
  const float* m2   = (const float*)d_in[14];
  const float* va2  = (const float*)d_in[15];
  float* out = (float*)d_out;

  char* ws = (char*)d_ws;
  half_t* wqk = (half_t*)ws; ws += (size_t)2 * DIM * DIM * sizeof(half_t);
  half_t* wv  = (half_t*)ws; ws += (size_t)DIM * DIM * sizeof(half_t);
  half_t* wpj = (half_t*)ws; ws += (size_t)DIM * DIM * sizeof(half_t);
  float* bn1s = (float*)ws;  ws += DIM * sizeof(float);
  float* bn1h = (float*)ws;  ws += DIM * sizeof(float);
  float* bn2s = (float*)ws;  ws += DIM * sizeof(float);
  float* bn2h = (float*)ws;  ws += DIM * sizeof(float);
  const size_t qkvN = (size_t)NWIN * HEADS * WS2 * HD;
  half_t* qbuf = (half_t*)ws; ws += qkvN * sizeof(half_t);
  half_t* kbuf = (half_t*)ws; ws += qkvN * sizeof(half_t);
  half_t* vbuf = (half_t*)ws; ws += qkvN * sizeof(half_t);
  half_t* ao   = (half_t*)ws; ws += (size_t)NWIN * WS2 * DIM * sizeof(half_t);

  prep_kernel<<<4100, 256, 0, stream>>>(qk_w, v_w, pw, g1, be1, m1, va1,
                                        g2, be2, m2, va2,
                                        wqk, wv, wpj, bn1s, bn1h, bn2s, bn2h);
  qkv_kernel<<<NWIN * 12, 256, 0, stream>>>(x, wqk, wv, bn1s, bn1h, qk_b, v_b,
                                            qbuf, kbuf, vbuf);
  attn_kernel<<<NWIN * HEADS, 128, 0, stream>>>(qbuf, kbuf, vbuf, bt, ao);
  proj_kernel<<<NWIN * 4, 256, 0, stream>>>(x, ao, wpj, pb, bn2s, bn2h, out);
}
